// Net_v2_58360015618510
// MI455X (gfx1250) — compile-verified, matrix operand order checked by reading, not run on hardware
//
#include <hip/hip_runtime.h>
#include <hip/hip_bf16.h>
#include <math.h>

// Problem constants (match reference)
#define T_STEPS 512
#define BSZ     64
#define INP     512
#define HS      2048
#define OUT     512
#define KC      64          // K-chunk staged per barrier

typedef __attribute__((ext_vector_type(8)))  float  v8f;
typedef __attribute__((ext_vector_type(16))) __bf16 v16bf;

union Frag {
    v16bf v;
    unsigned int u[8];
};

// ---------------------------------------------------------------------------
// CDNA5 data-movement feature detection (guarded so compile never regresses)
// ---------------------------------------------------------------------------
#if __has_builtin(__builtin_amdgcn_global_load_async_to_lds_b128)
#define HAVE_ASYNC_LDS 1
#else
#define HAVE_ASYNC_LDS 0
#endif

#if __has_builtin(__builtin_amdgcn_tensor_load_to_lds)
#define HAVE_TDM 1
#else
#define HAVE_TDM 0
#endif

// b128 async copy operand types: int __vector(4) in AS1 (global) / AS3 (LDS),
// matching the builtin's signature reported by the compiler.
typedef int i32x4v __attribute__((vector_size(16)));
typedef __attribute__((address_space(1))) i32x4v* gptr_b128;
typedef __attribute__((address_space(3))) i32x4v* lptr_b128;
typedef __attribute__((address_space(3))) char   las3_char;

__device__ __forceinline__ unsigned lds_byte_off(void* p) {
    // generic->LDS addrspace cast, then ptrtoint: yields byte offset in LDS
    return (unsigned)(unsigned long long)(las3_char*)p;
}

__device__ __forceinline__ void wait_asynccnt0() {
#if __has_builtin(__builtin_amdgcn_s_wait_asynccnt)
    __builtin_amdgcn_s_wait_asynccnt(0);
#else
    asm volatile("s_wait_asynccnt 0x0" ::: "memory");
#endif
}

#if HAVE_TDM
typedef unsigned int u32x4 __attribute__((ext_vector_type(4)));
typedef int          i32x4 __attribute__((ext_vector_type(4)));
typedef int          i32x8 __attribute__((ext_vector_type(8)));

// Issue one 2D TDM tile load: tileRows x tileK bf16 elements, row stride
// strideElems, into LDS at ldsOff (row-major, contiguous per row).
__device__ __forceinline__ void tdm_load_tile(const __bf16* gsrc, unsigned ldsOff,
                                              int tileK, int tileRows,
                                              long long strideElems)
{
    unsigned long long ga = (unsigned long long)gsrc;
    unsigned long long s0 = (unsigned long long)strideElems;   // data_size units
    unsigned td0 = (unsigned)tileK;        // tensor_dim0 (no OOB clipping needed)
    unsigned td1 = 0x100000u;              // large tensor_dim1

    u32x4 g0;
    g0[0] = 1u;                                            // count=1, user D#
    g0[1] = ldsOff;                                        // lds_addr (bytes)
    g0[2] = (unsigned)(ga & 0xFFFFFFFFull);                // global_addr[31:0]
    g0[3] = (unsigned)((ga >> 32) & 0x01FFFFFFull)         // global_addr[56:32]
          | 0x80000000u;                                   // type=2 ("image")

    i32x8 g1;
    g1[0] = (int)(1u << 16);                               // data_size=1 -> 2B
    g1[1] = (int)((td0 & 0xFFFFu) << 16);                  // tensor_dim0[15:0]
    g1[2] = (int)(((td0 >> 16) & 0xFFFFu) | ((td1 & 0xFFFFu) << 16));
    g1[3] = (int)(((td1 >> 16) & 0xFFFFu) | ((unsigned)tileK << 16));  // tile_dim0
    g1[4] = (int)((unsigned)tileRows & 0xFFFFu);           // tile_dim1 (tile_dim2=0)
    g1[5] = (int)(unsigned)(s0 & 0xFFFFFFFFull);           // dim0_stride[31:0]
    g1[6] = (int)(unsigned)((s0 >> 32) & 0xFFFFull);       // dim0_stride[47:32]
    g1[7] = 0;

    i32x4 z4 = {0, 0, 0, 0};
#if __has_include(<hip/amd_detail/amd_gfx1250_TDM.h>)
    i32x8 z8 = {0, 0, 0, 0, 0, 0, 0, 0};                   // therock: 6-arg builtin
    __builtin_amdgcn_tensor_load_to_lds(g0, g1, z4, z4, z8, 0);
#else
    __builtin_amdgcn_tensor_load_to_lds(g0, g1, z4, z4, 0); // ROCm 7.2: 5-arg
#endif
}
#endif // HAVE_TDM

// ---------------------------------------------------------------------------
// Shared GEMM main loop:  acc[4] (+)= A[64 x K] * B(k,n), B(k,n) = Bw[n][k]
//  - A  : bf16 row-major, lda elems/row (64 rows = batch)
//  - Bw : bf16 row-major [N][K], ldb elems/row; rows nBase..nBase+63
//  - block = 128 threads (4 waves); wave w owns N columns [nBase+16w, +16)
//  - acc[mt] = 16x16 f32 C/D tile for M rows [16*mt, +16)
// ---------------------------------------------------------------------------
__device__ __forceinline__ void gemm_phase(
    const __bf16* __restrict__ A, int lda,
    const __bf16* __restrict__ Bw, int ldb,
    int K, int nBase,
    __bf16 (*As)[KC], __bf16 (*Bs)[KC],
    v8f* acc)
{
    const int tid  = threadIdx.x;
    const int lane = tid & 31;          // wave32
    const int wave = tid >> 5;
    const int kb   = (lane < 16) ? 0 : 8;
    const int r16  = lane & 15;

#if HAVE_TDM
    const unsigned bsOff = lds_byte_off(&Bs[0][0]);
#endif

    for (int kc = 0; kc < K; kc += KC) {
        __syncthreads();   // previous chunk fully consumed before restaging

#if HAVE_TDM
        if (tid < 32) {    // one TDM descriptor per chunk, issued by wave 0
            tdm_load_tile(Bw + (size_t)nBase * ldb + kc, bsOff, KC, 64, ldb);
        }
#endif
        // Stage A[0..63][kc..kc+KC) (and Bw tile too when no TDM):
        // 64 rows x 128B = 512 x 16B chunks; 128 threads -> 4 chunks each.
#pragma unroll
        for (int i = 0; i < 4; ++i) {
            int idx  = tid + i * 128;
            int row  = idx >> 3;
            int part = idx & 7;
            const __bf16* ga = A + (size_t)row * lda + kc + part * 8;
#if HAVE_ASYNC_LDS
            __builtin_amdgcn_global_load_async_to_lds_b128(
                (gptr_b128)ga, (lptr_b128)(void*)&As[row][part * 8], 0, 0);
#else
            *(uint4*)(&As[row][part * 8]) = *(const uint4*)ga;
#endif
#if !HAVE_TDM
            const __bf16* gb = Bw + (size_t)(nBase + row) * ldb + kc + part * 8;
#if HAVE_ASYNC_LDS
            __builtin_amdgcn_global_load_async_to_lds_b128(
                (gptr_b128)gb, (lptr_b128)(void*)&Bs[row][part * 8], 0, 0);
#else
            *(uint4*)(&Bs[row][part * 8]) = *(const uint4*)gb;
#endif
#endif
            if (kc + KC < K) {   // warm next chunk in WGP$/L2 (global_prefetch_b8)
                __builtin_prefetch(ga + KC, 0, 1);
            }
        }
#if HAVE_ASYNC_LDS
        wait_asynccnt0();
#endif
#if HAVE_TDM
        if (tid < 32) __builtin_amdgcn_s_wait_tensorcnt(0);
#endif
        __syncthreads();

        // Two 32-wide sub-chunks -> 8 WMMAs per wave per barrier.
#pragma unroll
        for (int ks = 0; ks < KC; ks += 32) {
            Frag bf;
            const __bf16* brow = &Bs[wave * 16 + r16][ks];
#pragma unroll
            for (int i = 0; i < 4; ++i) {
                bf.u[i]     = *(const unsigned int*)(brow + kb + 2 * i);
                bf.u[4 + i] = *(const unsigned int*)(brow + 16 + kb + 2 * i);
            }
#pragma unroll
            for (int mt = 0; mt < 4; ++mt) {
                Frag af;
                const __bf16* arow = &As[mt * 16 + r16][ks];
#pragma unroll
                for (int i = 0; i < 4; ++i) {
                    af.u[i]     = *(const unsigned int*)(arow + kb + 2 * i);
                    af.u[4 + i] = *(const unsigned int*)(arow + 16 + kb + 2 * i);
                }
                acc[mt] = __builtin_amdgcn_wmma_f32_16x16x32_bf16(
                    false, af.v, false, bf.v, (short)0, acc[mt], false, false);
            }
        }
    }
}

// ---------------------------------------------------------------------------
// Step kernel A: r = tanh(x @ Wih^T + h @ Whh^T + bih + bhh)   [64 x HS]
// ---------------------------------------------------------------------------
__global__ void rnn_cell_kernel(const __bf16* __restrict__ xt,
                                const __bf16* __restrict__ h,
                                const __bf16* __restrict__ Wih,
                                const __bf16* __restrict__ Whh,
                                const float* __restrict__ bih,
                                const float* __restrict__ bhh,
                                __bf16* __restrict__ r)
{
    __shared__ __align__(16) __bf16 As[64][KC];
    __shared__ __align__(16) __bf16 Bs[64][KC];
    v8f acc[4];
    v8f z = {};
#pragma unroll
    for (int i = 0; i < 4; ++i) acc[i] = z;

    const int nBase = blockIdx.x * 64;
    gemm_phase(xt, INP, Wih, INP, INP, nBase, As, Bs, acc);
    gemm_phase(h,  HS,  Whh, HS,  HS,  nBase, As, Bs, acc);

    const int lane  = threadIdx.x & 31;
    const int wave  = threadIdx.x >> 5;
    const int n     = nBase + wave * 16 + (lane & 15);
    const int mhalf = (lane < 16) ? 0 : 8;
    const float bias = bih[n] + bhh[n];
#pragma unroll
    for (int mt = 0; mt < 4; ++mt) {
#pragma unroll
        for (int v = 0; v < 8; ++v) {
            int m = mt * 16 + mhalf + v;
            float val = acc[mt][v] + bias;
            r[(size_t)m * HS + n] = (__bf16)tanhf(val);
        }
    }
}

// ---------------------------------------------------------------------------
// Step kernel B: p = pot + r @ Wp^T + bp ; act = relu(p) ; pot = (p-act)*decay
// ---------------------------------------------------------------------------
__global__ void pglu_kernel(const __bf16* __restrict__ r,
                            const __bf16* __restrict__ Wp,
                            const float* __restrict__ bp,
                            const float* __restrict__ decay,
                            float* __restrict__ pot,
                            __bf16* __restrict__ h)
{
    __shared__ __align__(16) __bf16 As[64][KC];
    __shared__ __align__(16) __bf16 Bs[64][KC];
    v8f acc[4];
    v8f z = {};
#pragma unroll
    for (int i = 0; i < 4; ++i) acc[i] = z;

    const int nBase = blockIdx.x * 64;
    gemm_phase(r, HS, Wp, HS, HS, nBase, As, Bs, acc);

    const int lane  = threadIdx.x & 31;
    const int wave  = threadIdx.x >> 5;
    const int n     = nBase + wave * 16 + (lane & 15);
    const int mhalf = (lane < 16) ? 0 : 8;
    const float bpn = bp[n];
    const float dn  = decay[n];
#pragma unroll
    for (int mt = 0; mt < 4; ++mt) {
#pragma unroll
        for (int v = 0; v < 8; ++v) {
            int m = mt * 16 + mhalf + v;
            size_t idx = (size_t)m * HS + n;
            float p = acc[mt][v] + pot[idx] + bpn;
            float a = fmaxf(p, 0.0f);
            pot[idx] = (p - a) * dn;
            h[idx]   = (__bf16)a;
        }
    }
}

// ---------------------------------------------------------------------------
// Output projection: out = act_last @ Wout^T + bout   [64 x OUT], fp32 out
// ---------------------------------------------------------------------------
__global__ void out_proj_kernel(const __bf16* __restrict__ h,
                                const __bf16* __restrict__ Wout,
                                const float* __restrict__ bout,
                                float* __restrict__ out)
{
    __shared__ __align__(16) __bf16 As[64][KC];
    __shared__ __align__(16) __bf16 Bs[64][KC];
    v8f acc[4];
    v8f z = {};
#pragma unroll
    for (int i = 0; i < 4; ++i) acc[i] = z;

    const int nBase = blockIdx.x * 64;
    gemm_phase(h, HS, Wout, HS, HS, nBase, As, Bs, acc);

    const int lane  = threadIdx.x & 31;
    const int wave  = threadIdx.x >> 5;
    const int n     = nBase + wave * 16 + (lane & 15);
    const int mhalf = (lane < 16) ? 0 : 8;
    const float bn  = bout[n];
#pragma unroll
    for (int mt = 0; mt < 4; ++mt) {
#pragma unroll
        for (int v = 0; v < 8; ++v) {
            int m = mt * 16 + mhalf + v;
            out[(size_t)m * OUT + n] = acc[mt][v] + bn;
        }
    }
}

// ---------------------------------------------------------------------------
// Prep kernels
// ---------------------------------------------------------------------------
__global__ void cvt_f32_to_bf16(const float* __restrict__ src,
                                __bf16* __restrict__ dst, int n)
{
    int i = blockIdx.x * blockDim.x + threadIdx.x;
    if (i < n) dst[i] = (__bf16)src[i];
}

__global__ void zero_state_kernel(__bf16* __restrict__ h, float* __restrict__ pot, int n)
{
    int i = blockIdx.x * blockDim.x + threadIdx.x;
    if (i < n) {
        h[i]   = (__bf16)0.0f;
        pot[i] = 0.0f;
    }
}

// ---------------------------------------------------------------------------
// Workspace layout (bytes)
// ---------------------------------------------------------------------------
static const size_t OFF_WIH  = 0;                                        // 2*HS*INP
static const size_t OFF_WHH  = OFF_WIH  + (size_t)2 * HS * INP;          // 2*HS*HS
static const size_t OFF_WP   = OFF_WHH  + (size_t)2 * HS * HS;           // 2*HS*HS
static const size_t OFF_WOUT = OFF_WP   + (size_t)2 * HS * HS;           // 2*OUT*HS
static const size_t OFF_X    = OFF_WOUT + (size_t)2 * OUT * HS;          // 2*T*B*INP
static const size_t OFF_H    = OFF_X    + (size_t)2 * T_STEPS * BSZ * INP;
static const size_t OFF_R    = OFF_H    + (size_t)2 * BSZ * HS;
static const size_t OFF_POT  = OFF_R    + (size_t)2 * BSZ * HS;          // 4*B*HS

extern "C" void kernel_launch(void* const* d_in, const int* in_sizes, int n_in,
                              void* d_out, int out_size, void* d_ws, size_t ws_size,
                              hipStream_t stream)
{
    (void)in_sizes; (void)n_in; (void)out_size; (void)ws_size;

    const float* data  = (const float*)d_in[0];
    const float* Wih   = (const float*)d_in[1];
    const float* bih   = (const float*)d_in[2];
    const float* Whh   = (const float*)d_in[3];
    const float* bhh   = (const float*)d_in[4];
    const float* Wp    = (const float*)d_in[5];
    const float* bp    = (const float*)d_in[6];
    const float* decay = (const float*)d_in[7];
    const float* Wout  = (const float*)d_in[8];
    const float* bout  = (const float*)d_in[9];
    float* out = (float*)d_out;

    char* ws = (char*)d_ws;
    __bf16* Wih_bf  = (__bf16*)(ws + OFF_WIH);
    __bf16* Whh_bf  = (__bf16*)(ws + OFF_WHH);
    __bf16* Wp_bf   = (__bf16*)(ws + OFF_WP);
    __bf16* Wout_bf = (__bf16*)(ws + OFF_WOUT);
    __bf16* X_bf    = (__bf16*)(ws + OFF_X);
    __bf16* h_bf    = (__bf16*)(ws + OFF_H);
    __bf16* r_bf    = (__bf16*)(ws + OFF_R);
    float*  pot     = (float*)(ws + OFF_POT);

    // Per-call bf16 conversion of weights and input sequence.
    {
        int n;
        n = HS * INP;
        cvt_f32_to_bf16<<<(n + 255) / 256, 256, 0, stream>>>(Wih, Wih_bf, n);
        n = HS * HS;
        cvt_f32_to_bf16<<<(n + 255) / 256, 256, 0, stream>>>(Whh, Whh_bf, n);
        cvt_f32_to_bf16<<<(n + 255) / 256, 256, 0, stream>>>(Wp, Wp_bf, n);
        n = OUT * HS;
        cvt_f32_to_bf16<<<(n + 255) / 256, 256, 0, stream>>>(Wout, Wout_bf, n);
        n = T_STEPS * BSZ * INP;
        cvt_f32_to_bf16<<<(n + 255) / 256, 256, 0, stream>>>(data, X_bf, n);
        n = BSZ * HS;
        zero_state_kernel<<<(n + 255) / 256, 256, 0, stream>>>(h_bf, pot, n);
    }

    // Sequential scan: stream ordering provides the inter-kernel dependency.
    for (int t = 0; t < T_STEPS; ++t) {
        const __bf16* xt = X_bf + (size_t)t * BSZ * INP;
        rnn_cell_kernel<<<HS / 64, 128, 0, stream>>>(xt, h_bf, Wih_bf, Whh_bf,
                                                     bih, bhh, r_bf);
        pglu_kernel<<<HS / 64, 128, 0, stream>>>(r_bf, Wp_bf, bp, decay, pot, h_bf);
    }

    out_proj_kernel<<<OUT / 64, 128, 0, stream>>>(h_bf, Wout_bf, bout, out);
}
